// Matcher_4887672782910
// MI455X (gfx1250) — matrix operand order repaired and checked by reading, not compiled, not run on hardware
//
#include <hip/hip_runtime.h>
#include <hip/hip_bf16.h>

// CDNA5 / gfx1250: wave32, WMMA f32_16x16x32_bf16 path.
// fp32 GEMM emulated as bf16 hi/lo split: a*b ~= ah*bh + ah*bl + al*bh.
// hi/lo split precomputed ONCE into d_ws. GEMM: 64 cols/wave (4 WMMA tiles),
// B kept in LDS (compiler barrier stops register hoisting -> low VGPR count,
// high occupancy), A double-buffered + cross-tile prefetched, n-range split
// across waves/blocks with float atomicMax combine.

typedef __attribute__((ext_vector_type(16))) __bf16 v16bf;
typedef __attribute__((ext_vector_type(8)))  __bf16 v8bf;
typedef __attribute__((ext_vector_type(8)))  float  v8f;

#define B_    4
#define C_    256
#define HW_   2304
#define OUTC_ 10
#define NT_   144                       // HW_/16 row tiles total
#define NTW_  36                        // n-tiles per wave (NT_/4 splits)
#define COLS_ 64                        // columns per block/wave
#define TIL_  4                         // 16-col WMMA tiles per wave
#define P_    ((size_t)B_ * HW_ * C_)   // elements per plane (2,359,296)

union V16U { v16bf v; v8bf h[2]; };
union BFB  { unsigned short u; __bf16 b; };

__device__ __forceinline__ unsigned short f2bf_rne(float f) {
    unsigned int u = __float_as_uint(f);
    u += 0x7FFFu + ((u >> 16) & 1u);          // round-to-nearest-even
    return (unsigned short)(u >> 16);
}

__device__ __forceinline__ void splitf(float f, __bf16& hi, __bf16& lo) {
    unsigned short hb = f2bf_rne(f);
    float hf = __uint_as_float(((unsigned int)hb) << 16);
    BFB ch; ch.u = hb; hi = ch.b;
    BFB cl; cl.u = f2bf_rne(f - hf); lo = cl.b;
}

// ---------------------------------------------------------------------------
// Pass 1: elementwise fp32 -> bf16 hi/lo split of the 4 mem tensors and k.
// ws plane layout (bf16): [src 0..4][hi, lo][P_], src 4 = norm_key.
// grid = (P_/256, 5), block = 256.
// ---------------------------------------------------------------------------
__global__ __launch_bounds__(256)
void split_kernel(const float* __restrict__ g0, const float* __restrict__ g1,
                  const float* __restrict__ g2, const float* __restrict__ g3,
                  const float* __restrict__ kk, __bf16* __restrict__ ws)
{
    const int z = blockIdx.y;
    const float* src = (z == 0) ? g0 : (z == 1) ? g1 : (z == 2) ? g2
                     : (z == 3) ? g3 : kk;
    __bf16* hi = ws + (size_t)z * 2 * P_;
    __bf16* lo = hi + P_;

    const size_t i = (size_t)blockIdx.x * 256 + threadIdx.x;
    __bf16 h, l;
    splitf(src[i], h, l);
    hi[i] = h;
    lo[i] = l;
}

// ---------------------------------------------------------------------------
// Init fine channels (0..3 per batch) to -INF for the atomicMax combine.
// grid = (B_*4*HW_/256), block = 256.
// ---------------------------------------------------------------------------
__global__ __launch_bounds__(256)
void init_fine_kernel(float* __restrict__ out)
{
    const int i = blockIdx.x * 256 + threadIdx.x;   // over B_*4*HW_
    const int b = i / (4 * HW_);
    const int r = i - b * (4 * HW_);                // ch*HW_ + m, ch<4
    out[(size_t)b * OUTC_ * HW_ + r] = -INFINITY;
}

// ---------------------------------------------------------------------------
// Pass 2: fused GEMM + (sds mul) + column-max (atomic combine).
// grid = (36, 4, 2): 36 column-blocks of 64, 4 batches, 2 n-splits.
// block = 256 (8 waves): wave w -> mem (w&3), n-half (w>>2).
// mems: 0 global_bg->ch0, 1 global_fg->ch1, 2 local_bg*sds->ch2,
//       3 local_fg*sds->ch3.
// ---------------------------------------------------------------------------
__global__ __launch_bounds__(256)
void fine_max_kernel(const __bf16* __restrict__ ws,   // pre-split planes
                     const float*  __restrict__ sds,  // (B, HW, HW)
                     float* __restrict__ out)         // (B, 10, HW)
{
    __shared__ __bf16 sBh[COLS_ * C_];   // 32 KB: k cols, bf16 hi
    __shared__ __bf16 sBl[COLS_ * C_];   // 32 KB: k cols, bf16 lo

    const int b   = blockIdx.y;
    const int m0  = blockIdx.x * COLS_;
    const int tid = threadIdx.x;

    const __bf16* kHi = ws + (size_t)4 * 2 * P_;
    const __bf16* kLo = kHi + P_;

    // Stage B operand (k columns m0..m0+63, all 256 channels) into LDS.
    // Layout: sB[col*256 + c] -> contiguous K runs per column.
    for (int idx = tid; idx < COLS_ * C_; idx += 256) {
        const int col = idx & (COLS_ - 1);
        const int c   = idx >> 6;
        const size_t gi = ((size_t)b * C_ + c) * HW_ + (m0 + col);
        sBh[col * C_ + c] = kHi[gi];
        sBl[col * C_ + c] = kLo[gi];
    }
    __syncthreads();

    const int wave = tid >> 5;
    const int wm   = wave & 3;      // mem tensor index
    const int half = wave >> 2;     // n-range half within this block
    const int lane = tid & 31;
    const int row  = lane & 15;     // A: M row / B,D: N col
    const int g    = lane >> 4;     // half-wave group

    const __bf16* memHi = ws + (size_t)wm * 2 * P_;
    const __bf16* memLo = memHi + P_;
    const bool use_sds = (wm >= 2);

    // n-tile range for this wave: 36 tiles.
    const int ntBase = (blockIdx.z * 2 + half) * NTW_;

    // Per-lane base into A: row `row` of the current n-tile.
    const __bf16* paH = memHi + ((size_t)b * HW_ + row) * C_ + g * 8;
    const __bf16* paL = memLo + ((size_t)b * HW_ + row) * C_ + g * 8;

    float vmax[TIL_];
    #pragma unroll
    for (int t = 0; t < TIL_; ++t) vmax[t] = -INFINITY;

    // Double-buffered A chunk (hi/lo), pipelined across the n-loop.
    V16U ah0, al0, ah1, al1;
    auto loadA = [&](int nt_, int kc_, V16U& ah, V16U& al) {
        const size_t o = (size_t)nt_ * 16 * C_ + kc_ * 32;
        ah.h[0] = *(const v8bf*)(paH + o);
        ah.h[1] = *(const v8bf*)(paH + o + 16);
        al.h[0] = *(const v8bf*)(paL + o);
        al.h[1] = *(const v8bf*)(paL + o + 16);
    };

    loadA(ntBase, 0, ah0, al0);                  // prologue

    for (int i = 0; i < NTW_; ++i) {
        const int nt = ntBase + i;
        const int n0 = nt * 16;

        // Compiler barrier: keep the B operand in LDS (re-load each tile)
        // instead of hoisting ~128 VGPRs of loop-invariant B registers.
        asm volatile("" ::: "memory");

        v8f acc[TIL_];
        #pragma unroll
        for (int t = 0; t < TIL_; ++t) acc[t] = v8f{};

        #pragma unroll
        for (int kc = 0; kc < 8; ++kc) {
            V16U& ahC = (kc & 1) ? ah1 : ah0;    // current chunk
            V16U& alC = (kc & 1) ? al1 : al0;
            V16U& ahN = (kc & 1) ? ah0 : ah1;    // next chunk buffer
            V16U& alN = (kc & 1) ? al0 : al1;

            // Prefetch next K-chunk; at kc==7 prefetch chunk 0 of the next
            // n-tile (wrap to ntBase at the end: in-bounds, result unused).
            const int nnt = (kc == 7) ? ((i + 1 == NTW_) ? ntBase : nt + 1) : nt;
            const int nkc = (kc + 1) & 7;
            loadA(nnt, nkc, ahN, alN);

            #pragma unroll
            for (int t = 0; t < TIL_; ++t) {
                // B tile 32x16: lane = col N, elems j<8 -> K = g*8 + j, j>=8 -> K = 16 + g*8 + (j-8).
                const __bf16* pbh = &sBh[(t * 16 + row) * C_ + kc * 32 + g * 8];
                const __bf16* pbl = &sBl[(t * 16 + row) * C_ + kc * 32 + g * 8];
                V16U bh, bl;
                bh.h[0] = *(const v8bf*)(pbh);
                bh.h[1] = *(const v8bf*)(pbh + 16);
                bl.h[0] = *(const v8bf*)(pbl);
                bl.h[1] = *(const v8bf*)(pbl + 16);

                acc[t] = __builtin_amdgcn_wmma_f32_16x16x32_bf16(
                          false, ahC.v, false, bh.v, (short)0, acc[t], false, false);
                acc[t] = __builtin_amdgcn_wmma_f32_16x16x32_bf16(
                          false, ahC.v, false, bl.v, (short)0, acc[t], false, false);
                acc[t] = __builtin_amdgcn_wmma_f32_16x16x32_bf16(
                          false, alC.v, false, bh.v, (short)0, acc[t], false, false);
            }
        }

        // Fuse sds multiply + column max. D layout: elem r -> M = r + 8*g,
        // col N = row(lane&15). n = n0 + M, m = m0 + t*16 + N.
        if (use_sds) {
            #pragma unroll
            for (int t = 0; t < TIL_; ++t) {
                const float* ps = sds + ((size_t)b * HW_ + (n0 + g * 8)) * HW_
                                      + (m0 + t * 16 + row);
                #pragma unroll
                for (int r = 0; r < 8; ++r) {
                    // sds is a one-shot 85MB stream: non-temporal load.
                    const float w = __builtin_nontemporal_load(ps + (size_t)r * HW_);
                    vmax[t] = fmaxf(vmax[t], acc[t][r] * w);
                }
            }
        } else {
            #pragma unroll
            for (int t = 0; t < TIL_; ++t)
                #pragma unroll
                for (int r = 0; r < 8; ++r)
                    vmax[t] = fmaxf(vmax[t], acc[t][r]);
        }
    }

    // Combine the two M-halves (lanes N and N+16 hold the same column),
    // then merge across waves/blocks with float atomic max.
    #pragma unroll
    for (int t = 0; t < TIL_; ++t) {
        vmax[t] = fmaxf(vmax[t], __shfl_xor(vmax[t], 16, 32));
        if (g == 0) {
            float* po = out + ((size_t)b * OUTC_ + wm) * HW_ + m0 + t * 16 + row;
            atomicMax(po, vmax[t]);
        }
    }
}

// ---------------------------------------------------------------------------
// Coarse channels 4..9: out[b, 4+ch, m] = sum_c mem[b,0,c] * k[b,c,m].
// grid = (9, 6, 4), block = 256 (HW = 9*256).
// ---------------------------------------------------------------------------
__global__ __launch_bounds__(256)
void coarse_kernel(const float* __restrict__ k,
                   const float* __restrict__ ob, const float* __restrict__ of,
                   const float* __restrict__ sb, const float* __restrict__ sf,
                   const float* __restrict__ lb, const float* __restrict__ lf,
                   float* __restrict__ out)
{
    const int m  = blockIdx.x * 256 + threadIdx.x;
    const int ch = blockIdx.y;
    const int b  = blockIdx.z;

    const float* mp;
    switch (ch) {
        case 0:  mp = ob; break;
        case 1:  mp = of; break;
        case 2:  mp = sb; break;
        case 3:  mp = sf; break;
        case 4:  mp = lb; break;
        default: mp = lf; break;
    }
    mp += (size_t)b * C_;

    const float* kp = k + (size_t)b * C_ * HW_ + m;
    float acc = 0.f;
    #pragma unroll 8
    for (int c = 0; c < C_; ++c)
        acc = fmaf(mp[c], kp[(size_t)c * HW_], acc);

    out[((size_t)b * OUTC_ + 4 + ch) * HW_ + m] = acc;
}

extern "C" void kernel_launch(void* const* d_in, const int* in_sizes, int n_in,
                              void* d_out, int out_size, void* d_ws, size_t ws_size,
                              hipStream_t stream) {
    (void)in_sizes; (void)n_in; (void)out_size; (void)ws_size;
    const float* k   = (const float*)d_in[0];   // norm_key  (B,C,H,W)
    const float* gbg = (const float*)d_in[1];   // global_bg (B,HW,C)
    const float* gfg = (const float*)d_in[2];   // global_fg
    const float* lbg = (const float*)d_in[3];   // local_bg
    const float* lfg = (const float*)d_in[4];   // local_fg
    const float* ob  = (const float*)d_in[5];   // overall_bg (B,1,C)
    const float* of  = (const float*)d_in[6];   // overall_fg
    const float* sb  = (const float*)d_in[7];   // short_bg
    const float* sf  = (const float*)d_in[8];   // short_fg
    const float* lb  = (const float*)d_in[9];   // long_bg
    const float* lf  = (const float*)d_in[10];  // long_fg
    const float* sds = (const float*)d_in[11];  // (B,HW,H,W)
    float* out   = (float*)d_out;
    __bf16* wsbf = (__bf16*)d_ws;               // needs 10*P_*2 bytes (~47 MB)

    // Pass 1: precompute bf16 hi/lo planes for the 4 mems and k.
    dim3 gridS((unsigned)(P_ / 256), 5, 1);
    split_kernel<<<gridS, dim3(256, 1, 1), 0, stream>>>(
        gbg, gfg, lbg, lfg, k, wsbf);

    // Init fine channels for atomic max combine.
    init_fine_kernel<<<dim3((B_ * 4 * HW_) / 256, 1, 1), dim3(256, 1, 1),
                       0, stream>>>(out);

    // Pass 2: WMMA GEMM + fused sds/max, atomic combine.
    dim3 gridF(HW_ / COLS_, B_, 2);             // 36 x 4 x 2
    fine_max_kernel<<<gridF, dim3(256, 1, 1), 0, stream>>>(
        wsbf, sds, out);

    // Coarse channels.
    dim3 gridC(HW_ / 256, 6, B_);               // 9 x 6 x 4
    coarse_kernel<<<gridC, dim3(256, 1, 1), 0, stream>>>(
        k, ob, of, sb, sf, lb, lf, out);
}
